// FEPSurrogateNetwork_73993696575528
// MI455X (gfx1250) — matrix operand order repaired and checked by reading, not compiled, not run on hardware
//
#include <hip/hip_runtime.h>
#include <math.h>

#define N_NODES 50000
#define N_EDGES 600000
#define NPAD    50048   // 391*128 = 782*64 : padded node rows, guard-free kernels
#define EPAD    600064  // 4688*128        : padded edge count
#define HD 128
#define NLAYERS 4
#define EIN 385
#define ASTR 136  // padded activation-tile stride (elements): 272B, breaks 256B bank aliasing

typedef __attribute__((ext_vector_type(16))) __bf16 v16bf;
typedef __attribute__((ext_vector_type(8)))  __bf16 v8bf;
typedef __attribute__((ext_vector_type(8)))  float  v8f;

struct bf16x16 { v8bf lo, hi; };

__device__ __forceinline__ float silu_f(float x) { return x / (1.0f + __expf(-x)); }
__device__ __forceinline__ __bf16 bfc(float x) { return (__bf16)x; }

// ---- A fragment (16x32 MxK bf16), row-major LDS tile, 2x ds_load_b128 ----
__device__ __forceinline__ v16bf frag_a(const __bf16* p, int stride) {
  const int lane = threadIdx.x & 31;
  const int half = lane >> 4;
  const int row  = lane & 15;
  const __bf16* base = p + row * stride + half * 8;
  bf16x16 t;
  t.lo = *(const v8bf*)(base);
  t.hi = *(const v8bf*)(base + 16);
  return __builtin_bit_cast(v16bf, t);
}

// ---- B fragment from pre-packed LDS: [frag][lane][16] contiguous, 2x ds_load_b128 ----
__device__ __forceinline__ v16bf frag_b(const __bf16* fragbase) {
  const int lane = threadIdx.x & 31;
  const __bf16* base = fragbase + lane * 16;
  bf16x16 t;
  t.lo = *(const v8bf*)(base);
  t.hi = *(const v8bf*)(base + 8);
  return __builtin_bit_cast(v16bf, t);
}

// Pack a 128x128 row-major (KxN) f32 weight into per-fragment bf16 layout.
__device__ __forceinline__ void pack_b(__bf16* dst, const float* __restrict__ W,
                                       int tid, int nthreads) {
  for (int j = tid; j < HD * HD; j += nthreads) {
    int k = j >> 7, n = j & 127;
    int kt = k >> 5, kr = k & 31, half = (kr >> 3) & 1, i = (kr & 7) + ((kr >> 4) << 3);
    int nt = n >> 4, col = n & 15;
    dst[((kt * 8 + nt) * 32 + (col + half * 16)) * 16 + i] = bfc(W[j]);
  }
}
// C/D layout (16x16 f32): elem r -> row (r + 8*half), col (lane&15)

// ---------------- init (covers all NPAD rows; z gather clamped) ----------------
__global__ void k_init(const int* __restrict__ z, const float* __restrict__ atom_embed,
                       const float* __restrict__ lam_p, const float* __restrict__ lam_w,
                       const float* __restrict__ lam_b, float* __restrict__ h,
                       float* __restrict__ agg) {
  int idx = blockIdx.x * blockDim.x + threadIdx.x;   // one float4 per thread
  if (idx >= NPAD * HD / 4) return;
  int n = idx >> 5, d0 = (idx & 31) * 4;
  int zn = n < N_NODES ? n : N_NODES - 1;
  float lam = lam_p[0];
  float4 ae = *(const float4*)(atom_embed + z[zn] * HD + d0);
  float4 lw = *(const float4*)(lam_w + d0);
  float4 lb = *(const float4*)(lam_b + d0);
  float4 hv;
  hv.x = ae.x + lam * lw.x + lb.x;
  hv.y = ae.y + lam * lw.y + lb.y;
  hv.z = ae.z + lam * lw.z + lb.z;
  hv.w = ae.w + lam * lw.w + lb.w;
  *(float4*)(h + (size_t)idx * 4)   = hv;
  *(float4*)(agg + (size_t)idx * 4) = float4{0.f, 0.f, 0.f, 0.f};
}

// ---------------- per-edge distances (covers EPAD; ei reads clamped) ----------------
__global__ void k_dist(const float* __restrict__ pos, const int* __restrict__ ei,
                       float* __restrict__ dist) {
  int e = blockIdx.x * blockDim.x + threadIdx.x;
  if (e >= EPAD) return;
  int ec = e < N_EDGES ? e : N_EDGES - 1;
  int r = ei[ec], c = ei[N_EDGES + ec];
  float dx = pos[r * 3 + 0] - pos[c * 3 + 0];
  float dy = pos[r * 3 + 1] - pos[c * 3 + 1];
  float dz = pos[r * 3 + 2] - pos[c * 3 + 2];
  dist[e] = sqrtf(dx * dx + dy * dy + dz * dz + 1e-8f);
}

// ---------------- c_lam[l] = lam_vec @ edge_w1[l][257:385] + edge_b1[l] ----------------
__global__ void k_clam(const float* __restrict__ lam_p, const float* __restrict__ lam_w,
                       const float* __restrict__ lam_b, const float* __restrict__ edge_w1,
                       const float* __restrict__ edge_b1, float* __restrict__ clam) {
  int idx = blockIdx.x * blockDim.x + threadIdx.x;
  if (idx >= NLAYERS * HD) return;
  int i = idx >> 7, d = idx & 127;
  float lam = lam_p[0];
  const float* w = edge_w1 + (size_t)i * EIN * HD + 257 * HD + d;
  float s = edge_b1[i * HD + d];
  for (int k = 0; k < HD; ++k) s += (lam * lam_w[k] + lam_b[k]) * w[k * HD];
  clam[idx] = s;
}

// ---------------- fused projections: C1 = A@B1, C2 = A@B2 (A staged once, guard-free) ----------------
__global__ void k_gemm_proj2(const float* __restrict__ A,
                             const float* __restrict__ B1, const float* __restrict__ B2,
                             float* __restrict__ C1, float* __restrict__ C2) {
  extern __shared__ char smem[];
  __bf16* Bp1 = (__bf16*)smem;                         // packed 128x128
  __bf16* Bp2 = Bp1 + HD * HD;                         // packed 128x128
  __bf16* As  = Bp2 + HD * HD;                         // 8 waves x 16xASTR
  const int tid = threadIdx.x, wave = tid >> 5, lane = tid & 31;
  pack_b(Bp1, B1, tid, 256);
  pack_b(Bp2, B2, tid, 256);
  const int r0 = blockIdx.x * 128 + wave * 16;
  __bf16* Aw = As + wave * 16 * ASTR;
  for (int it = 0; it < 8; ++it) {
    int c = it * 32 + lane;
    int r = c >> 4, d0 = (c & 15) * 8;
    const float* ap = A + (size_t)(r0 + r) * HD + d0;
    float4 a0 = *(const float4*)(ap);
    float4 a1 = *(const float4*)(ap + 4);
    v8bf o;
    o[0]=bfc(a0.x); o[1]=bfc(a0.y); o[2]=bfc(a0.z); o[3]=bfc(a0.w);
    o[4]=bfc(a1.x); o[5]=bfc(a1.y); o[6]=bfc(a1.z); o[7]=bfc(a1.w);
    *(v8bf*)(Aw + r * ASTR + d0) = o;
  }
  __syncthreads();
  const int half = lane >> 4, col = lane & 15;
#pragma unroll
  for (int sel = 0; sel < 2; ++sel) {
    const __bf16* Bp = sel ? Bp2 : Bp1;
    float* C = sel ? C2 : C1;
#pragma unroll
    for (int nt = 0; nt < 8; ++nt) {
      v8f acc = {};
#pragma unroll
      for (int kt = 0; kt < 4; ++kt) {
        v16bf a = frag_a(Aw + kt * 32, ASTR);
        v16bf b = frag_b(Bp + (kt * 8 + nt) * 512);
        acc = __builtin_amdgcn_wmma_f32_16x16x32_bf16(false, a, false, b, (short)0, acc, false, false);
      }
      float* cbase = C + (size_t)(r0 + half * 8) * HD + nt * 16 + col;
#pragma unroll
      for (int r = 0; r < 8; ++r) cbase[(size_t)r * HD] = acc[r];
    }
  }
}

// ---------------- edge kernel (guard-free; padded edges scatter to sink row N_NODES) ----------------
__global__ void k_edge(const float* __restrict__ Pa, const float* __restrict__ Pb,
                       const float* __restrict__ dist, const int* __restrict__ ei,
                       const float* __restrict__ wd, const float* __restrict__ clam_l,
                       const float* __restrict__ W2g, const float* __restrict__ b2g,
                       float* __restrict__ agg) {
  extern __shared__ char smem[];
  __bf16* W2p = (__bf16*)smem;                                   // packed 128x128
  __bf16* M1  = (__bf16*)(smem + HD * HD * 2);                   // 8 x 16xASTR
  int*    ROW = (int*)(smem + HD * HD * 2 + 8 * 16 * ASTR * 2);  // 8 x 16
  int*    COL = ROW + 8 * 16;
  const int tid = threadIdx.x, wave = tid >> 5, lane = tid & 31;
  pack_b(W2p, W2g, tid, 256);
  const int e0 = blockIdx.x * 128 + wave * 16;
  int* rw = ROW + wave * 16;
  int* cw = COL + wave * 16;
  if (lane < 16) {
    int e = e0 + lane;
    bool in = e < N_EDGES;
    rw[lane] = in ? ei[e] : N_NODES;             // sink row for padded edges
    cw[lane] = in ? ei[N_EDGES + e] : N_NODES;
  }
  __syncthreads();
  __bf16* m1 = M1 + wave * 16 * ASTR;
  for (int it = 0; it < 8; ++it) {
    int c = it * 32 + lane;
    int r = c >> 4, d0 = (c & 15) * 8;
    float de = dist[e0 + r];
    const float* pa = Pa + (size_t)rw[r] * HD + d0;
    const float* pb = Pb + (size_t)cw[r] * HD + d0;
    float4 a0 = *(const float4*)(pa),     a1 = *(const float4*)(pa + 4);
    float4 b0 = *(const float4*)(pb),     b1 = *(const float4*)(pb + 4);
    float4 w0 = *(const float4*)(wd + d0),      w1 = *(const float4*)(wd + d0 + 4);
    float4 c0 = *(const float4*)(clam_l + d0),  c1 = *(const float4*)(clam_l + d0 + 4);
    v8bf o;
    o[0]=bfc(silu_f(a0.x+b0.x+de*w0.x+c0.x)); o[1]=bfc(silu_f(a0.y+b0.y+de*w0.y+c0.y));
    o[2]=bfc(silu_f(a0.z+b0.z+de*w0.z+c0.z)); o[3]=bfc(silu_f(a0.w+b0.w+de*w0.w+c0.w));
    o[4]=bfc(silu_f(a1.x+b1.x+de*w1.x+c1.x)); o[5]=bfc(silu_f(a1.y+b1.y+de*w1.y+c1.y));
    o[6]=bfc(silu_f(a1.z+b1.z+de*w1.z+c1.z)); o[7]=bfc(silu_f(a1.w+b1.w+de*w1.w+c1.w));
    *(v8bf*)(m1 + r * ASTR + d0) = o;
  }
  __syncthreads();
  const int half = lane >> 4, col = lane & 15;
#pragma unroll
  for (int nt = 0; nt < 8; ++nt) {
    const float b2 = b2g[nt * 16 + col];
    v8f acc;
#pragma unroll
    for (int r = 0; r < 8; ++r) acc[r] = b2;   // bias folded into accumulator
#pragma unroll
    for (int kt = 0; kt < 4; ++kt) {
      v16bf a = frag_a(m1 + kt * 32, ASTR);
      v16bf b = frag_b(W2p + (kt * 8 + nt) * 512);
      acc = __builtin_amdgcn_wmma_f32_16x16x32_bf16(false, a, false, b, (short)0, acc, false, false);
    }
#pragma unroll
    for (int r = 0; r < 8; ++r) {
      int m = r + half * 8;
      atomicAdd(&agg[(size_t)rw[m] * HD + nt * 16 + col], silu_f(acc[r]));
    }
  }
}

// ---------------- node kernel (guard-free over NPAD) ----------------
__global__ void k_node(float* __restrict__ h, float* __restrict__ agg,
                       const float* __restrict__ W1g, const float* __restrict__ b1g,
                       const float* __restrict__ W2g, const float* __restrict__ b2g,
                       const float* __restrict__ g_l, const float* __restrict__ be_l) {
  extern __shared__ char smem[];
  __bf16* W1a = (__bf16*)smem;              // packed 128x128 (h part)
  __bf16* W1b = W1a + HD * HD;              // packed 128x128 (agg part)
  __bf16* W2p = W1b + HD * HD;              // packed 128x128
  __bf16* HT  = W2p + HD * HD;              // 4 x 16xASTR
  __bf16* GT  = HT + 4 * 16 * ASTR;
  __bf16* U1  = GT + 4 * 16 * ASTR;
  float*  XS  = (float*)(U1 + 4 * 16 * ASTR); // 4 x 16x128 f32
  const int tid = threadIdx.x, wave = tid >> 5, lane = tid & 31;
  pack_b(W1a, W1g,           tid, 128);
  pack_b(W1b, W1g + HD * HD, tid, 128);
  pack_b(W2p, W2g,           tid, 128);
  const int n0 = blockIdx.x * 64 + wave * 16;
  __bf16* ht = HT + wave * 16 * ASTR;
  __bf16* gt = GT + wave * 16 * ASTR;
  __bf16* u1 = U1 + wave * 16 * ASTR;
  float*  xs = XS + wave * 16 * HD;
  for (int it = 0; it < 8; ++it) {
    int c = it * 32 + lane;
    int r = c >> 4, d0 = (c & 15) * 8;
    size_t base = (size_t)(n0 + r) * HD + d0;
    float4 h0 = *(const float4*)(h + base);   float4 h1 = *(const float4*)(h + base + 4);
    float4 g0 = *(const float4*)(agg + base); float4 g1 = *(const float4*)(agg + base + 4);
    *(float4*)(agg + base)     = float4{0.f,0.f,0.f,0.f};  // reset for next layer
    *(float4*)(agg + base + 4) = float4{0.f,0.f,0.f,0.f};
    v8bf oh, og;
    oh[0]=bfc(h0.x); oh[1]=bfc(h0.y); oh[2]=bfc(h0.z); oh[3]=bfc(h0.w);
    oh[4]=bfc(h1.x); oh[5]=bfc(h1.y); oh[6]=bfc(h1.z); oh[7]=bfc(h1.w);
    og[0]=bfc(g0.x); og[1]=bfc(g0.y); og[2]=bfc(g0.z); og[3]=bfc(g0.w);
    og[4]=bfc(g1.x); og[5]=bfc(g1.y); og[6]=bfc(g1.z); og[7]=bfc(g1.w);
    *(v8bf*)(ht + r * ASTR + d0) = oh;
    *(v8bf*)(gt + r * ASTR + d0) = og;
  }
  __syncthreads();
  const int half = lane >> 4, col = lane & 15;
  // u1 = silu([h,agg] @ W1 + b1)
#pragma unroll
  for (int nt = 0; nt < 8; ++nt) {
    const float b1 = b1g[nt * 16 + col];
    v8f acc;
#pragma unroll
    for (int r = 0; r < 8; ++r) acc[r] = b1;
#pragma unroll
    for (int kt = 0; kt < 4; ++kt) {
      v16bf a = frag_a(ht + kt * 32, ASTR);
      v16bf b = frag_b(W1a + (kt * 8 + nt) * 512);
      acc = __builtin_amdgcn_wmma_f32_16x16x32_bf16(false, a, false, b, (short)0, acc, false, false);
    }
#pragma unroll
    for (int kt = 0; kt < 4; ++kt) {
      v16bf a = frag_a(gt + kt * 32, ASTR);
      v16bf b = frag_b(W1b + (kt * 8 + nt) * 512);
      acc = __builtin_amdgcn_wmma_f32_16x16x32_bf16(false, a, false, b, (short)0, acc, false, false);
    }
#pragma unroll
    for (int r = 0; r < 8; ++r) {
      int m = r + half * 8;
      u1[m * ASTR + nt * 16 + col] = bfc(silu_f(acc[r]));
    }
  }
  __syncthreads();
  // x = h + u1 @ W2 + b2
#pragma unroll
  for (int nt = 0; nt < 8; ++nt) {
    const float b2 = b2g[nt * 16 + col];
    v8f acc;
#pragma unroll
    for (int r = 0; r < 8; ++r) acc[r] = b2;
#pragma unroll
    for (int kt = 0; kt < 4; ++kt) {
      v16bf a = frag_a(u1 + kt * 32, ASTR);
      v16bf b = frag_b(W2p + (kt * 8 + nt) * 512);
      acc = __builtin_amdgcn_wmma_f32_16x16x32_bf16(false, a, false, b, (short)0, acc, false, false);
    }
#pragma unroll
    for (int r = 0; r < 8; ++r) {
      int m = r + half * 8;
      float hv = h[(size_t)(n0 + m) * HD + nt * 16 + col];  // f32 residual
      xs[m * HD + nt * 16 + col] = hv + acc[r];
    }
  }
  __syncthreads();
  // LayerNorm per row (wave32 shuffle reductions)
  for (int r = 0; r < 16; ++r) {
    float s = 0.0f;
#pragma unroll
    for (int q = 0; q < 4; ++q) s += xs[r * HD + lane + q * 32];
#pragma unroll
    for (int m = 16; m >= 1; m >>= 1) s += __shfl_xor(s, m, 32);
    float mu = s * (1.0f / HD);
    float v = 0.0f;
#pragma unroll
    for (int q = 0; q < 4; ++q) { float t = xs[r * HD + lane + q * 32] - mu; v += t * t; }
#pragma unroll
    for (int m = 16; m >= 1; m >>= 1) v += __shfl_xor(v, m, 32);
    float rs = rsqrtf(v * (1.0f / HD) + 1e-5f);
#pragma unroll
    for (int q = 0; q < 4; ++q) {
      int cc = lane + q * 32;
      h[(size_t)(n0 + r) * HD + cc] = (xs[r * HD + cc] - mu) * rs * g_l[cc] + be_l[cc];
    }
  }
}

// ---------------- pooling + head ----------------
__global__ void k_zero_pool(float* __restrict__ pooled) {
  if (threadIdx.x < HD) pooled[threadIdx.x] = 0.0f;
}
__global__ void k_pool(const float* __restrict__ h, float* __restrict__ pooled) {
  const int d = threadIdx.x;  // 128 threads
  const int chunk = (N_NODES + gridDim.x - 1) / gridDim.x;
  const int s0 = blockIdx.x * chunk;
  int s1 = s0 + chunk; if (s1 > N_NODES) s1 = N_NODES;
  float s = 0.0f;
  for (int n = s0; n < s1; ++n) s += h[(size_t)n * HD + d];
  atomicAdd(&pooled[d], s);
}
__global__ void k_head(const float* __restrict__ pooled, const float* __restrict__ W1,
                       const float* __restrict__ b1, const float* __restrict__ W2,
                       const float* __restrict__ b2, float* __restrict__ out) {
  __shared__ float x[HD];
  __shared__ float t[HD];
  int j = threadIdx.x;
  x[j] = pooled[j] * (1.0f / N_NODES);
  __syncthreads();
  float s = b1[j];
  for (int d = 0; d < HD; ++d) s += x[d] * W1[d * HD + j];
  t[j] = silu_f(s) * W2[j];
  __syncthreads();
  if (j == 0) {
    float o = b2[0];
    for (int d = 0; d < HD; ++d) o += t[d];
    out[0] = o;
  }
}

extern "C" void kernel_launch(void* const* d_in, const int* in_sizes, int n_in,
                              void* d_out, int out_size, void* d_ws, size_t ws_size,
                              hipStream_t stream) {
  (void)in_sizes; (void)n_in; (void)out_size; (void)ws_size;
  const int*   z          = (const int*)  d_in[0];
  const float* pos        = (const float*)d_in[1];
  const int*   ei         = (const int*)  d_in[2];
  const float* lam        = (const float*)d_in[3];
  const float* atom_embed = (const float*)d_in[4];
  const float* lam_w      = (const float*)d_in[5];
  const float* lam_b      = (const float*)d_in[6];
  const float* edge_w1    = (const float*)d_in[7];
  const float* edge_b1    = (const float*)d_in[8];
  const float* edge_w2    = (const float*)d_in[9];
  const float* edge_b2    = (const float*)d_in[10];
  const float* node_w1    = (const float*)d_in[11];
  const float* node_b1    = (const float*)d_in[12];
  const float* node_w2    = (const float*)d_in[13];
  const float* node_b2    = (const float*)d_in[14];
  const float* ln_g       = (const float*)d_in[15];
  const float* ln_b       = (const float*)d_in[16];
  const float* head_w1    = (const float*)d_in[17];
  const float* head_b1    = (const float*)d_in[18];
  const float* head_w2    = (const float*)d_in[19];
  const float* head_b2    = (const float*)d_in[20];
  float* out = (float*)d_out;

  char* ws = (char*)d_ws;
  size_t off = 0;
  float* h      = (float*)(ws + off); off += (size_t)NPAD * HD * 4;
  float* agg    = (float*)(ws + off); off += (size_t)NPAD * HD * 4;
  float* Pa     = (float*)(ws + off); off += (size_t)NPAD * HD * 4;
  float* Pb     = (float*)(ws + off); off += (size_t)NPAD * HD * 4;
  float* dist   = (float*)(ws + off); off += (size_t)EPAD * 4;
  float* clam   = (float*)(ws + off); off += (size_t)NLAYERS * HD * 4;
  float* pooled = (float*)(ws + off); off += HD * 4;

  k_init<<<(NPAD * HD / 4 + 255) / 256, 256, 0, stream>>>(z, atom_embed, lam, lam_w, lam_b, h, agg);
  k_dist<<<(EPAD + 255) / 256, 256, 0, stream>>>(pos, ei, dist);
  k_clam<<<(NLAYERS * HD + 127) / 128, 128, 0, stream>>>(lam, lam_w, lam_b, edge_w1, edge_b1, clam);

  const size_t proj_lds = (size_t)2 * HD * HD * 2 + 8 * 16 * ASTR * 2;
  const size_t edge_lds = (size_t)HD * HD * 2 + 8 * 16 * ASTR * 2 + 2 * 8 * 16 * 4;
  const size_t node_lds = (size_t)3 * HD * HD * 2 + 3 * 4 * 16 * ASTR * 2 + 4 * 16 * HD * 4;

  const int proj_grid = NPAD / 128;   // 391
  const int edge_grid = EPAD / 128;   // 4688
  const int node_grid = NPAD / 64;    // 782

  for (int l = 0; l < NLAYERS; ++l) {
    const float* w1l = edge_w1 + (size_t)l * EIN * HD;
    k_gemm_proj2<<<proj_grid, 256, proj_lds, stream>>>(h, w1l, w1l + 128 * HD, Pa, Pb);
    k_edge<<<edge_grid, 256, edge_lds, stream>>>(Pa, Pb, dist, ei,
        w1l + 256 * HD, clam + l * HD,
        edge_w2 + (size_t)l * HD * HD, edge_b2 + l * HD, agg);
    k_node<<<node_grid, 128, node_lds, stream>>>(h, agg,
        node_w1 + (size_t)l * 2 * HD * HD, node_b1 + l * HD,
        node_w2 + (size_t)l * HD * HD, node_b2 + l * HD,
        ln_g + l * HD, ln_b + l * HD);
  }
  k_zero_pool<<<1, 128, 0, stream>>>(pooled);
  k_pool<<<128, 128, 0, stream>>>(h, pooled);
  k_head<<<1, 128, 0, stream>>>(pooled, head_w1, head_b1, head_w2, head_b2, out);
}
